// MultiheadAttentionMechanism_49546742726987
// MI455X (gfx1250) — matrix operand order repaired and checked by reading, not compiled, not run on hardware
//
#include <hip/hip_runtime.h>

#define BDIM 32
#define TLEN 1500
#define DDIM 512
#define ADIM 512
#define HNUM 4
#define CNUM 10
#define KHALF 100
#define KLEN 201

typedef __attribute__((ext_vector_type(16))) __bf16 v16bf;
typedef __attribute__((ext_vector_type(8))) unsigned int v8u;
typedef __attribute__((ext_vector_type(16))) unsigned short v16us;
typedef __attribute__((ext_vector_type(8))) float v8f;

// Branch-free round-to-nearest-even f32 -> bf16 (inputs are finite).
__device__ __forceinline__ unsigned int bfround(float f) {
    unsigned int u = __float_as_uint(f);
    return u + 0x7fffu + ((u >> 16) & 1u);
}
// Pack two f32 into one dword of two bf16 (lo in [15:0]) with one v_perm_b32.
__device__ __forceinline__ unsigned int pack2bf(float lo, float hi) {
    return __builtin_amdgcn_perm(bfround(hi), bfround(lo), 0x07060302u);
}
__device__ __forceinline__ unsigned short f2bf(float f) {
    return (unsigned short)(bfround(f) >> 16);
}

// Hardware transcendental tanh (TRANS32). v_nop satisfies the CDNA5 rule:
// one independent op between a trans op and first use of its result.
__device__ __forceinline__ float fast_tanh(float x) {
    float r;
    asm volatile("v_tanh_f32 %0, %1\n\tv_nop" : "=v"(r) : "v"(x));
    return r;
}

// ---------------------------------------------------------------------------
// Pack w_enc_W [512,512] f32 (k-major rows) into per-lane WMMA B-operand bf16
// layout: tile(kt,nt), lane holds col n = nt*16 + (lane&15),
// K = kt*32 + ((lane&16)?16:0) + j, j = 0..15, 16 bf16 contiguous per lane.
// ---------------------------------------------------------------------------
__global__ void pack_wenc_kernel(const float* __restrict__ W,
                                 unsigned short* __restrict__ packW) {
    int idx = blockIdx.x * 256 + threadIdx.x;      // 16 kt * 32 nt * 32 lanes
    int kt   = idx >> 10;
    int nt   = (idx >> 5) & 31;
    int lane = idx & 31;
    int n = nt * 16 + (lane & 15);
    int kbase = kt * 32 + ((lane & 16) ? 16 : 0);
    unsigned short* out = packW + (size_t)idx * 16;
#pragma unroll
    for (int j = 0; j < 16; ++j)
        out[j] = f2bf(W[(size_t)(kbase + j) * ADIM + n]);
}

__global__ void zero_kernel(float* __restrict__ p, int n) {
    int i = blockIdx.x * 256 + threadIdx.x;
    if (i < n) p[i] = 0.f;
}

// ---------------------------------------------------------------------------
// dec[b,a] = sum_d query[b,d] * w_dec_W[d,a]
// ---------------------------------------------------------------------------
__global__ void dec_kernel(const float* __restrict__ query,
                           const float* __restrict__ Wd,
                           float* __restrict__ dec) {
    int b = blockIdx.x >> 1;
    int a = ((blockIdx.x & 1) << 8) + threadIdx.x;
    __shared__ float q_s[DDIM];
    for (int i = threadIdx.x; i < DDIM; i += 256) q_s[i] = query[(size_t)b * DDIM + i];
    __syncthreads();
    float acc = 0.f;
    for (int d = 0; d < DDIM; ++d) acc += q_s[d] * Wd[(size_t)d * ADIM + a];
    dec[(size_t)b * ADIM + a] = acc;
}

// ---------------------------------------------------------------------------
// Grouped 1D conv: feat[(h*B+b)*T+t][c] = sum_j aw[h,b,t-100+j] * convW[h*C+c][j]
// ---------------------------------------------------------------------------
__global__ void conv_kernel(const float* __restrict__ aw,
                            const float* __restrict__ convW,
                            float* __restrict__ feat) {
    int hb = blockIdx.y;                 // h*BDIM + b
    int h = hb >> 5;
    int t00 = blockIdx.x * 256;
    __shared__ float aw_s[256 + 2 * KHALF];
    __shared__ float k_s[CNUM][KLEN];
    for (int i = threadIdx.x; i < 256 + 2 * KHALF; i += 256) {
        int t = t00 - KHALF + i;
        aw_s[i] = (t >= 0 && t < TLEN) ? aw[(size_t)hb * TLEN + t] : 0.f;
    }
    for (int i = threadIdx.x; i < CNUM * KLEN; i += 256)
        k_s[i / KLEN][i % KLEN] = convW[(size_t)h * CNUM * KLEN + i];
    __syncthreads();
    int t = t00 + threadIdx.x;
    if (t < TLEN) {
        float* outp = feat + ((size_t)hb * TLEN + t) * CNUM;
#pragma unroll
        for (int c = 0; c < CNUM; ++c) {
            float s = 0.f;
            for (int j = 0; j < KLEN; ++j) s += aw_s[threadIdx.x + j] * k_s[c][j];
            outp[c] = s;
        }
    }
}

// ---------------------------------------------------------------------------
// Flagship: key_a GEMM (bf16 WMMA, f32 accum) fused with bias + dec +
// conv-projection + tanh + v-dot -> per-head scores e[h,b,t].
// Workgroup = 8 waves = 256 threads, handles (b, 16 t-rows). Wave w owns
// N columns [64w, 64w+64): 4 WMMA accum tiles. K = 512 in 16 steps of 32.
// ---------------------------------------------------------------------------
__global__ void __launch_bounds__(256)
score_gemm_kernel(const float* __restrict__ key,
                  const unsigned short* __restrict__ packW,
                  const float* __restrict__ enc_b,
                  const float* __restrict__ dec,
                  const float* __restrict__ feat,
                  const float* __restrict__ Wc,
                  const float* __restrict__ vW,
                  float* __restrict__ e_g) {
    const int tid  = threadIdx.x;
    const int wave = tid >> 5;
    const int lane = tid & 31;
    const int b  = blockIdx.y;
    const int t0 = blockIdx.x * 16;

    __shared__ float feat_s[HNUM][16][CNUM];
    __shared__ float wc_s[CNUM][ADIM];
    __shared__ float dec_s[ADIM];
    __shared__ float encb_s[ADIM];
    __shared__ float vw_s[ADIM];
    __shared__ float e_s[HNUM][16];

    for (int i = tid; i < HNUM * 16 * CNUM; i += 256) {
        int h = i / (16 * CNUM), rem = i % (16 * CNUM);
        int m = rem / CNUM, c = rem % CNUM;
        int t = t0 + m; if (t >= TLEN) t = TLEN - 1;
        feat_s[h][m][c] = feat[(((size_t)h * BDIM + b) * TLEN + t) * CNUM + c];
    }
    for (int i = tid; i < CNUM * ADIM; i += 256)
        wc_s[i / ADIM][i % ADIM] = Wc[i];
    for (int i = tid; i < ADIM; i += 256) {
        dec_s[i]  = dec[(size_t)b * ADIM + i];
        encb_s[i] = enc_b[i];
        vw_s[i]   = vW[i];
    }
    if (tid < HNUM * 16) e_s[tid >> 4][tid & 15] = 0.f;
    __syncthreads();

    // ---- WMMA main loop ----
    int trow = t0 + (lane & 15); if (trow >= TLEN) trow = TLEN - 1;
    const float* keyrow = key + ((size_t)b * TLEN + trow) * DDIM;
    const int kb = (lane & 16) ? 16 : 0;   // A-matrix K half per lane group

    v8f acc[4];
#pragma unroll
    for (int i = 0; i < 4; ++i) acc[i] = (v8f)(0.f);

    for (int kt = 0; kt < 16; ++kt) {
        int k0 = kt * 32 + kb;
        float4 a0 = *(const float4*)(keyrow + k0);
        float4 a1 = *(const float4*)(keyrow + k0 + 4);
        float4 a2 = *(const float4*)(keyrow + k0 + 8);
        float4 a3 = *(const float4*)(keyrow + k0 + 12);
        v8u au;
        au[0] = pack2bf(a0.x, a0.y);
        au[1] = pack2bf(a0.z, a0.w);
        au[2] = pack2bf(a1.x, a1.y);
        au[3] = pack2bf(a1.z, a1.w);
        au[4] = pack2bf(a2.x, a2.y);
        au[5] = pack2bf(a2.z, a2.w);
        au[6] = pack2bf(a3.x, a3.y);
        au[7] = pack2bf(a3.z, a3.w);
        v16bf amat = __builtin_bit_cast(v16bf, au);
#pragma unroll
        for (int i = 0; i < 4; ++i) {
            int tileIdx = kt * 32 + (wave * 4 + i);
            const v16us* bp = (const v16us*)packW + (size_t)tileIdx * 32 + lane;
            v16bf bmat = __builtin_bit_cast(v16bf, *bp);
            acc[i] = __builtin_amdgcn_wmma_f32_16x16x32_bf16(
                false, amat, false, bmat, (short)0, acc[i], false, false);
        }
    }

    // ---- fused epilogue: + bias + dec + conv_proj, tanh, * v, reduce over n ----
    const int n_lo = wave * 64 + (lane & 15);
    float biasv[4], vwv[4], wcn[4][CNUM];
#pragma unroll
    for (int i = 0; i < 4; ++i) {
        int n = n_lo + i * 16;
        biasv[i] = encb_s[n] + dec_s[n];
        vwv[i]   = vw_s[n];
#pragma unroll
        for (int c = 0; c < CNUM; ++c) wcn[i][c] = wc_s[c][n];
    }

    float esum[HNUM][8];
#pragma unroll
    for (int h = 0; h < HNUM; ++h)
#pragma unroll
        for (int r = 0; r < 8; ++r) esum[h][r] = 0.f;

    const int mofs = (lane & 16) ? 8 : 0;
#pragma unroll
    for (int r = 0; r < 8; ++r) {
        int m = r + mofs;
        float fr[HNUM][CNUM];
#pragma unroll
        for (int h = 0; h < HNUM; ++h)
#pragma unroll
            for (int c = 0; c < CNUM; ++c) fr[h][c] = feat_s[h][m][c];
#pragma unroll
        for (int i = 0; i < 4; ++i) {
            float x0 = acc[i][r] + biasv[i];
#pragma unroll
            for (int h = 0; h < HNUM; ++h) {
                float cp = 0.f;
#pragma unroll
                for (int c = 0; c < CNUM; ++c) cp += fr[h][c] * wcn[i][c];
                esum[h][r] += fast_tanh(x0 + cp) * vwv[i];
            }
        }
    }

    // cross-lane sum over the 16 n-lanes in each half, then LDS accumulate
#pragma unroll
    for (int h = 0; h < HNUM; ++h) {
#pragma unroll
        for (int r = 0; r < 8; ++r) {
            float v = esum[h][r];
            v += __shfl_xor(v, 1);
            v += __shfl_xor(v, 2);
            v += __shfl_xor(v, 4);
            v += __shfl_xor(v, 8);
            if ((lane & 15) == 0) atomicAdd(&e_s[h][r + mofs], v);
        }
    }
    __syncthreads();

    if (tid < HNUM * 16) {
        int h = tid >> 4, m = tid & 15;
        int t = t0 + m;
        if (t < TLEN) e_g[((size_t)h * BDIM + b) * TLEN + t] = e_s[h][m];
    }
}

// ---------------------------------------------------------------------------
// Masked softmax over T per (h,b); writes aw_new.
// ---------------------------------------------------------------------------
__global__ void softmax_kernel(const float* __restrict__ e_g,
                               const int* __restrict__ key_lens,
                               float* __restrict__ aw_out) {
    int row = blockIdx.x;                 // h*BDIM + b
    int b = row & 31;
    int len = key_lens[b];
    const float* e = e_g + (size_t)row * TLEN;
    __shared__ float red[256];
    float m = -3.0e38f;
    for (int t = threadIdx.x; t < len; t += 256) m = fmaxf(m, e[t]);
    red[threadIdx.x] = m;
    __syncthreads();
    for (int s = 128; s > 0; s >>= 1) {
        if (threadIdx.x < s) red[threadIdx.x] = fmaxf(red[threadIdx.x], red[threadIdx.x + s]);
        __syncthreads();
    }
    m = red[0];
    __syncthreads();
    float sum = 0.f;
    for (int t = threadIdx.x; t < len; t += 256) sum += __expf(e[t] - m);
    red[threadIdx.x] = sum;
    __syncthreads();
    for (int s = 128; s > 0; s >>= 1) {
        if (threadIdx.x < s) red[threadIdx.x] += red[threadIdx.x + s];
        __syncthreads();
    }
    float inv = 1.f / red[0];
    for (int t = threadIdx.x; t < TLEN; t += 256)
        aw_out[(size_t)row * TLEN + t] = (t < len) ? __expf(e[t] - m) * inv : 0.f;
}

// ---------------------------------------------------------------------------
// ctx[b, h*D + v] += sum_t aw[h,b,t] * value[b,t,v]  (t split across blocks)
// ---------------------------------------------------------------------------
__global__ void ctx_kernel(const float* __restrict__ aw,
                           const float* __restrict__ value,
                           float* __restrict__ ctx) {
    int b = blockIdx.z;
    int v = (blockIdx.y << 8) + threadIdx.x;
    int tbase = blockIdx.x * 250;
    __shared__ float aw_s[HNUM][250];
    for (int i = threadIdx.x; i < HNUM * 250; i += 256) {
        int h = i / 250, tt = i % 250;
        aw_s[h][tt] = aw[((size_t)h * BDIM + b) * TLEN + tbase + tt];
    }
    __syncthreads();
    float a0 = 0.f, a1 = 0.f, a2 = 0.f, a3 = 0.f;
    const float* vp = value + ((size_t)b * TLEN + tbase) * DDIM + v;
    for (int tt = 0; tt < 250; ++tt) {
        float val = vp[(size_t)tt * DDIM];
        a0 += val * aw_s[0][tt];
        a1 += val * aw_s[1][tt];
        a2 += val * aw_s[2][tt];
        a3 += val * aw_s[3][tt];
    }
    atomicAdd(&ctx[((size_t)b * HNUM + 0) * DDIM + v], a0);
    atomicAdd(&ctx[((size_t)b * HNUM + 1) * DDIM + v], a1);
    atomicAdd(&ctx[((size_t)b * HNUM + 2) * DDIM + v], a2);
    atomicAdd(&ctx[((size_t)b * HNUM + 3) * DDIM + v], a3);
}

// ---------------------------------------------------------------------------
// out[b,d] = sum_k ctx[b,k] * Wo[k,d] + bo[d]
// ---------------------------------------------------------------------------
__global__ void out_kernel(const float* __restrict__ ctx,
                           const float* __restrict__ Wo,
                           const float* __restrict__ bo,
                           float* __restrict__ out) {
    int b = blockIdx.x >> 1;
    int dcol = ((blockIdx.x & 1) << 8) + threadIdx.x;
    __shared__ float c_s[HNUM * DDIM];
    for (int i = threadIdx.x; i < HNUM * DDIM; i += 256)
        c_s[i] = ctx[(size_t)b * HNUM * DDIM + i];
    __syncthreads();
    float acc = bo[dcol];
    for (int k = 0; k < HNUM * DDIM; ++k) acc += c_s[k] * Wo[(size_t)k * DDIM + dcol];
    out[(size_t)b * DDIM + dcol] = acc;
}

extern "C" void kernel_launch(void* const* d_in, const int* in_sizes, int n_in,
                              void* d_out, int out_size, void* d_ws, size_t ws_size,
                              hipStream_t stream) {
    const float* key      = (const float*)d_in[0];
    const int*   key_lens = (const int*)d_in[1];
    const float* value    = (const float*)d_in[2];
    const float* query    = (const float*)d_in[3];
    const float* aw       = (const float*)d_in[4];
    const float* w_enc_W  = (const float*)d_in[5];
    const float* w_enc_b  = (const float*)d_in[6];
    const float* w_dec_W  = (const float*)d_in[7];
    const float* w_conv_W = (const float*)d_in[8];
    const float* v_W      = (const float*)d_in[9];
    const float* conv_W   = (const float*)d_in[10];
    const float* w_out_W  = (const float*)d_in[11];
    const float* w_out_b  = (const float*)d_in[12];

    float* out    = (float*)d_out;                 // [B,1,D] = 16384 floats
    float* aw_new = (float*)d_out + BDIM * DDIM;   // [H,B,T] = 192000 floats

    char* ws = (char*)d_ws;
    unsigned short* packW = (unsigned short*)(ws + 0);          // 512 KB
    float* feat = (float*)(ws + 524288);                        // H*B*T*C f32 = 7.68 MB
    float* dec  = (float*)(ws + 8204544);                       // B*A f32
    float* e_g  = (float*)(ws + 8270080);                       // H*B*T f32
    float* ctx  = (float*)(ws + 9038336);                       // B*H*D f32

    pack_wenc_kernel<<<64, 256, 0, stream>>>(w_enc_W, packW);
    zero_kernel<<<(BDIM * HNUM * DDIM + 255) / 256, 256, 0, stream>>>(ctx, BDIM * HNUM * DDIM);
    dec_kernel<<<BDIM * 2, 256, 0, stream>>>(query, w_dec_W, dec);
    conv_kernel<<<dim3(6, HNUM * BDIM), 256, 0, stream>>>(aw, conv_W, feat);
    score_gemm_kernel<<<dim3((TLEN + 15) / 16, BDIM), 256, 0, stream>>>(
        key, packW, w_enc_b, dec, feat, w_conv_W, v_W, e_g);
    softmax_kernel<<<HNUM * BDIM, 256, 0, stream>>>(e_g, key_lens, aw_new);
    ctx_kernel<<<dim3(6, 2, BDIM), 256, 0, stream>>>(aw, value, ctx);
    out_kernel<<<BDIM * 2, 256, 0, stream>>>(ctx, w_out_W, w_out_b, out);
}
